// MultiHeadAttention_4612794876533
// MI455X (gfx1250) — matrix operand order repaired and checked by reading, not compiled, not run on hardware
//
#include <hip/hip_runtime.h>
#include <hip/hip_bf16.h>

#define CDIM 512
#define SDIM 2048
#define BDIM 2
#define HDIM 8

typedef _Float16 v8h  __attribute__((ext_vector_type(8)));
typedef _Float16 v16h __attribute__((ext_vector_type(16)));
typedef float    v8f  __attribute__((ext_vector_type(8)));

__device__ __forceinline__ v8f wmma16(v16h a, v16h b, v8f c) {
  return __builtin_amdgcn_wmma_f32_16x16x32_f16(false, a, false, b, (short)0, c,
                                                false, false);
}

// A-fragment 16x32 (MxK): element (m,k) at base[(row0+m)*ld + k0 + k].
// wave32 layout: lane<16 -> elems0-7 = K0..7, elems8-15 = K16..23 (hid*8 shift).
__device__ __forceinline__ v16h load_afrag(const _Float16* __restrict__ base,
                                           int ld, int row0, int k0, int lane) {
  const int hid = lane >> 4, lr = lane & 15;
  const _Float16* p = base + (size_t)(row0 + lr) * ld + k0 + hid * 8;
  v8h lo = *(const v8h*)p;
  v8h hi = *(const v8h*)(p + 16);
  v16h a;
#pragma unroll
  for (int e = 0; e < 8; ++e) { a[e] = lo[e]; a[8 + e] = hi[e]; }
  return a;
}

// B-fragment 32x16 (KxN): element (k,n) at base[(n0+n)*ld + k0 + k].
// wave32 layout: lane<16 -> elems0-7 = K0..7, elems8-15 = K16..23 (hid*8 shift).
__device__ __forceinline__ v16h load_bfrag(const _Float16* __restrict__ base,
                                           int ld, int k0, int n0, int lane) {
  const int hid = lane >> 4, lr = lane & 15;
  const _Float16* p = base + (size_t)(n0 + lr) * ld + k0 + hid * 8;
  v8h lo = *(const v8h*)p;
  v8h hi = *(const v8h*)(p + 16);
  v16h bfrag;
#pragma unroll
  for (int e = 0; e < 8; ++e) { bfrag[e] = lo[e]; bfrag[8 + e] = hi[e]; }
  return bfrag;
}

// ---- K0a: fp32 weights -> f16, packed [Wq|Wk|Wv|Wo] row-major ----
__global__ void k_cvt_w(const float* __restrict__ Wq, const float* __restrict__ Wk,
                        const float* __restrict__ Wv, const float* __restrict__ Wo,
                        _Float16* __restrict__ dst) {
  const size_t CC = (size_t)CDIM * CDIM;
  size_t i = (size_t)blockIdx.x * blockDim.x + threadIdx.x;
  if (i >= 4 * CC) return;
  int w = (int)(i / CC);
  size_t r = i - (size_t)w * CC;
  const float* src = (w == 0) ? Wq : (w == 1) ? Wk : (w == 2) ? Wv : Wo;
  dst[i] = (_Float16)src[r];
}

// ---- K0b: fp32 q/k/v (B,C,1,S) -> f16 transposed [proj][b][s][c] ----
__global__ void k_cvt_xT(const float* __restrict__ q, const float* __restrict__ k,
                         const float* __restrict__ v, _Float16* __restrict__ xT) {
  const size_t per = (size_t)BDIM * CDIM * SDIM;
  size_t i = (size_t)blockIdx.x * blockDim.x + threadIdx.x;
  if (i >= 3 * per) return;
  int proj = (int)(i / per);
  size_t rem = i - (size_t)proj * per;
  const float* src = (proj == 0) ? q : (proj == 1) ? k : v;
  int b = (int)(rem / ((size_t)CDIM * SDIM));
  size_t r2 = rem % ((size_t)CDIM * SDIM);
  int c = (int)(r2 / SDIM);
  int s = (int)(r2 % SDIM);
  xT[(size_t)proj * per + ((size_t)b * SDIM + s) * CDIM + c] = (_Float16)src[rem];
}

// ---- K1: fused QKV projection. Out = W * x + bias.
// Q/K stored per-head transposed [bh][s][64] (Q pre-scaled by dh^-0.5),
// V stored natural [b][c][s].
__global__ __launch_bounds__(128) void k_proj(
    const _Float16* __restrict__ Wh, const _Float16* __restrict__ xT,
    _Float16* __restrict__ qpT, _Float16* __restrict__ kpT,
    _Float16* __restrict__ vp, const float* __restrict__ bq,
    const float* __restrict__ bk, const float* __restrict__ bv) {
  const int z = blockIdx.z, proj = z >> 1, b = z & 1;
  const _Float16* W = Wh + (size_t)proj * CDIM * CDIM;
  const _Float16* X = xT + ((size_t)proj * BDIM + b) * (size_t)SDIM * CDIM;
  const float* bias = (proj == 0) ? bq : (proj == 1) ? bk : bv;
  const float mult = (proj == 0) ? 0.125f : 1.0f;  // 1/sqrt(64) folded into Q

  const int lane = threadIdx.x & 31, wave = threadIdx.x >> 5;
  const int hid = lane >> 4, lr = lane & 15;
  const int m0 = blockIdx.y * 64 + (wave >> 1) * 32;
  const int n0 = blockIdx.x * 64 + (wave & 1) * 32;

  v8f acc[2][2] = {};
  for (int k0 = 0; k0 < CDIM; k0 += 32) {
    v16h a0 = load_afrag(W, CDIM, m0, k0, lane);
    v16h a1 = load_afrag(W, CDIM, m0 + 16, k0, lane);
    v16h b0 = load_bfrag(X, CDIM, k0, n0, lane);
    v16h b1 = load_bfrag(X, CDIM, k0, n0 + 16, lane);
    acc[0][0] = wmma16(a0, b0, acc[0][0]);
    acc[0][1] = wmma16(a0, b1, acc[0][1]);
    acc[1][0] = wmma16(a1, b0, acc[1][0]);
    acc[1][1] = wmma16(a1, b1, acc[1][1]);
  }

#pragma unroll
  for (int i = 0; i < 2; ++i)
#pragma unroll
    for (int j = 0; j < 2; ++j)
#pragma unroll
      for (int r = 0; r < 8; ++r) {
        const int o = m0 + i * 16 + r + 8 * hid;   // D-frag row
        const int s = n0 + j * 16 + lr;            // D-frag col
        const float val = (acc[i][j][r] + bias[o]) * mult;
        if (proj == 2) {
          vp[((size_t)b * CDIM + o) * SDIM + s] = (_Float16)val;
        } else {
          _Float16* dst = (proj == 0) ? qpT : kpT;
          dst[(((size_t)(b * HDIM + (o >> 6))) * SDIM + s) * 64 + (o & 63)] =
              (_Float16)val;
        }
      }
}

// ---- K2: flash attention per (b,h). One wave = 16 query columns.
// scores D-frag (rows=kpos, cols=q) IS the A-frag layout of P^T -> no shuffle
// between the two GEMMs. Accumulates O^T[q][d], writes attnT [b][s][c].
__global__ __launch_bounds__(128) void k_attn(
    const _Float16* __restrict__ qpT, const _Float16* __restrict__ kpT,
    const _Float16* __restrict__ vpn, _Float16* __restrict__ attnT) {
  const int lane = threadIdx.x & 31, wave = threadIdx.x >> 5;
  const int hid = lane >> 4, lr = lane & 15;
  const int bh = blockIdx.y, b = bh >> 3, h = bh & 7;
  const int q0 = blockIdx.x * 64 + wave * 16;

  const _Float16* qb = qpT + (size_t)bh * SDIM * 64;
  const _Float16* kb = kpT + (size_t)bh * SDIM * 64;
  const _Float16* vb = vpn + ((size_t)b * CDIM + h * 64) * SDIM;

  // B-operand of scores GEMM: qp[d][q] from qpT[q][d]; loop-invariant.
  v16h qf0 = load_bfrag(qb, 64, 0, q0, lane);
  v16h qf1 = load_bfrag(qb, 64, 32, q0, lane);

  v8f accO[4] = {};                 // O^T frags: rows q(16), cols d (4x16)
  float m_run = -1e30f, l_run = 0.0f;   // per-lane for q = lr (both halves)

  for (int kblk = 0; kblk < SDIM; kblk += 32) {
    // scores S[kpos][q] = sum_d kp^T[kpos][d] * qp[d][q]  (scale already in qp)
    v8f s0 = {}, s1 = {};
    s0 = wmma16(load_afrag(kb, 64, kblk, 0, lane), qf0, s0);
    s0 = wmma16(load_afrag(kb, 64, kblk, 32, lane), qf1, s0);
    s1 = wmma16(load_afrag(kb, 64, kblk + 16, 0, lane), qf0, s1);
    s1 = wmma16(load_afrag(kb, 64, kblk + 16, 32, lane), qf1, s1);

    // online softmax over key axis; column q lives in lanes lr and lr+16
    float mloc = -1e30f;
#pragma unroll
    for (int r = 0; r < 8; ++r) {
      mloc = fmaxf(mloc, s0[r]);
      mloc = fmaxf(mloc, s1[r]);
    }
    mloc = fmaxf(mloc, __shfl_xor(mloc, 16, 32));
    const float new_m = fmaxf(m_run, mloc);
    const float alpha = __expf(m_run - new_m);

    float psum = 0.0f;
    v16h p;  // A-fragment of P^T (16 q x 32 kpos) -- pure per-lane convert
#pragma unroll
    for (int r = 0; r < 8; ++r) {
      const float p0 = __expf(s0[r] - new_m);
      const float p1 = __expf(s1[r] - new_m);
      psum += p0 + p1;
      p[r] = (_Float16)p0;
      p[8 + r] = (_Float16)p1;
    }
    psum += __shfl_xor(psum, 16, 32);
    l_run = l_run * alpha + psum;
    m_run = new_m;

    // rescale O^T rows (row q = r + 8*hid) by alpha of that q
#pragma unroll
    for (int r = 0; r < 8; ++r) {
      const float ar = __shfl(alpha, r + 8 * hid, 32);
      accO[0][r] *= ar; accO[1][r] *= ar; accO[2][r] *= ar; accO[3][r] *= ar;
    }
    // O^T[q][d] += P^T[q][kpos] * V^T[kpos][d]; V^T contiguous in vp[d][s]
#pragma unroll
    for (int nt = 0; nt < 4; ++nt) {
      v16h vf = load_bfrag(vb, SDIM, kblk, nt * 16, lane);
      accO[nt] = wmma16(p, vf, accO[nt]);
    }
  }

  const float linv = 1.0f / l_run;
#pragma unroll
  for (int r = 0; r < 8; ++r) {
    const float li = __shfl(linv, r + 8 * hid, 32);
    const int qg = q0 + r + 8 * hid;
#pragma unroll
    for (int nt = 0; nt < 4; ++nt) {
      attnT[((size_t)b * SDIM + qg) * CDIM + h * 64 + nt * 16 + lr] =
          (_Float16)(accO[nt][r] * li);
    }
  }
}

// ---- K3: output projection, fp32 store to (B,C,1,S) ----
__global__ __launch_bounds__(128) void k_out(
    const _Float16* __restrict__ WhO, const _Float16* __restrict__ attnT,
    const float* __restrict__ bo, float* __restrict__ out) {
  const int b = blockIdx.z;
  const _Float16* X = attnT + (size_t)b * SDIM * CDIM;
  const int lane = threadIdx.x & 31, wave = threadIdx.x >> 5;
  const int hid = lane >> 4, lr = lane & 15;
  const int m0 = blockIdx.y * 64 + (wave >> 1) * 32;
  const int n0 = blockIdx.x * 64 + (wave & 1) * 32;

  v8f acc[2][2] = {};
  for (int k0 = 0; k0 < CDIM; k0 += 32) {
    v16h a0 = load_afrag(WhO, CDIM, m0, k0, lane);
    v16h a1 = load_afrag(WhO, CDIM, m0 + 16, k0, lane);
    v16h b0 = load_bfrag(X, CDIM, k0, n0, lane);
    v16h b1 = load_bfrag(X, CDIM, k0, n0 + 16, lane);
    acc[0][0] = wmma16(a0, b0, acc[0][0]);
    acc[0][1] = wmma16(a0, b1, acc[0][1]);
    acc[1][0] = wmma16(a1, b0, acc[1][0]);
    acc[1][1] = wmma16(a1, b1, acc[1][1]);
  }
#pragma unroll
  for (int i = 0; i < 2; ++i)
#pragma unroll
    for (int j = 0; j < 2; ++j)
#pragma unroll
      for (int r = 0; r < 8; ++r) {
        const int o = m0 + i * 16 + r + 8 * hid;
        const int s = n0 + j * 16 + lr;
        out[((size_t)b * CDIM + o) * SDIM + s] = acc[i][j][r] + bo[o];
      }
}

extern "C" void kernel_launch(void* const* d_in, const int* in_sizes, int n_in,
                              void* d_out, int out_size, void* d_ws, size_t ws_size,
                              hipStream_t stream) {
  const float* q  = (const float*)d_in[0];
  const float* k  = (const float*)d_in[1];
  const float* v  = (const float*)d_in[2];
  const float* Wq = (const float*)d_in[3];
  const float* bq = (const float*)d_in[4];
  const float* Wk = (const float*)d_in[5];
  const float* bk = (const float*)d_in[6];
  const float* Wv = (const float*)d_in[7];
  const float* bv = (const float*)d_in[8];
  const float* Wo = (const float*)d_in[9];
  const float* bo = (const float*)d_in[10];
  float* out = (float*)d_out;

  const size_t CC  = (size_t)CDIM * CDIM;          // 262144
  const size_t BSC = (size_t)BDIM * SDIM * CDIM;   // 2097152
  _Float16* ws    = (_Float16*)d_ws;
  _Float16* Wh    = ws;                 // 4*CC   [Wq|Wk|Wv|Wo] f16
  _Float16* xT    = ws + 4 * CC;        // 3*BSC  [proj][b][s][c]
  _Float16* qpT   = xT + 3 * BSC;       // BSC    [bh][s][64] (scaled)
  _Float16* kpT   = qpT + BSC;          // BSC    [bh][s][64]
  _Float16* vp    = kpT + BSC;          // BSC    [b][c][s]
  _Float16* attnT = vp + BSC;           // BSC    [b][s][c]

  k_cvt_w<<<dim3((unsigned)((4 * CC + 255) / 256)), dim3(256), 0, stream>>>(
      Wq, Wk, Wv, Wo, Wh);
  k_cvt_xT<<<dim3((unsigned)((3 * BSC + 255) / 256)), dim3(256), 0, stream>>>(
      q, k, v, xT);
  k_proj<<<dim3(SDIM / 64, CDIM / 64, 3 * BDIM), dim3(128), 0, stream>>>(
      Wh, xT, qpT, kpT, vp, bq, bk, bv);
  k_attn<<<dim3(SDIM / 64, BDIM * HDIM), dim3(128), 0, stream>>>(
      qpT, kpT, vp, attnT);
  k_out<<<dim3(SDIM / 64, CDIM / 64, BDIM), dim3(128), 0, stream>>>(
      Wh + 3 * CC, attnT, bo, out);
}
